// MatchingNet_65146063945875
// MI455X (gfx1250) — compile-verified, matrix operand order checked by reading, not run on hardware
//
#include <hip/hip_runtime.h>
#include <hip/hip_bf16.h>
#include <math.h>

#define DEVFN static __device__ __forceinline__

typedef __attribute__((ext_vector_type(16))) _Float16 v16h;
typedef __attribute__((ext_vector_type(8)))  float    v8f;
typedef __attribute__((ext_vector_type(4)))  unsigned v4u;
typedef __attribute__((ext_vector_type(4)))  int      v4i;
typedef __attribute__((ext_vector_type(8)))  int      v8i;

// ---------------- constants ----------------
#define BATCH 2
#define LSEQ 4096
#define DIM 256
#define HF 256
#define HW 65536            // 256*256
#define CF 128
#define WW 25
#define NMATCH 16384        // 2*B*L
#define MROWS 409600        // NMATCH*WW
#define CCHUNK 16           // column-scan parallel chunks

// d_out layout (floats): [local_desc | cm | j_ids | valid]
#define OUT_CM_OFF   ((size_t)52428800)           // 16384*25*128
#define OUT_J_OFF    (OUT_CM_OFF + (size_t)33554432)
#define OUT_V_OFF    (OUT_J_OFF + (size_t)8192)

// TDM: 5-arg builtin ships with ROCm (no TDM header); 6-arg with therock headers.
#if defined(__gfx1250__) && !__has_include(<hip/amd_detail/amd_gfx1250_TDM.h>)
#if __has_builtin(__builtin_amdgcn_tensor_load_to_lds)
#define USE_TDM 1
#endif
#endif

// ---------------- WMMA helpers ----------------
DEVFN v8f wmma_f16(v16h a, v16h b, v8f c) {
  return __builtin_amdgcn_wmma_f32_16x16x32_f16(false, a, false, b, (short)0, c,
                                                false, false);
}

// A/B fragment from a row-major [rows, ld] f16 matrix; tile origin given.
// 16-bit 16x32 layout: lane%16 = M/N, lane/16 selects K half-group,
// VGPR i holds K pair (i<4: 2i ; i>=4: 16+2(i-4)) + 8*(lane/16).
DEVFN v16h load_frag(const _Float16* __restrict__ tile, int ld, int lane) {
  const int r  = lane & 15;
  const int g8 = (lane >> 4) * 8;
  const _Float16* p = tile + (size_t)r * ld + g8;
  v16h f;
#pragma unroll
  for (int i = 0; i < 8; ++i) {
    const int k0 = (i < 4) ? (2 * i) : (16 + 2 * (i - 4));
    f[2 * i]     = p[k0];
    f[2 * i + 1] = p[k0 + 1];
  }
  return f;
}

// A fragment gathered from NCHW f32 featmap: element[m,k] = x[k*HW + (pix0+m)]
DEVFN v16h load_frag_nchw(const float* __restrict__ x, int pix0, int kbase, int lane) {
  const int m  = pix0 + (lane & 15);
  const int g8 = (lane >> 4) * 8;
  v16h f;
#pragma unroll
  for (int i = 0; i < 8; ++i) {
    const int k0 = ((i < 4) ? (2 * i) : (16 + 2 * (i - 4))) + g8 + kbase;
    f[2 * i]     = (_Float16)x[(size_t)k0 * HW + m];
    f[2 * i + 1] = (_Float16)x[(size_t)(k0 + 1) * HW + m];
  }
  return f;
}

// A fragment from a gathered row pointer (128 contiguous f16) or zeros.
DEVFN v16h load_frag_row(const _Float16* __restrict__ rowp, bool ok, int kbase, int lane) {
  v16h f = {};
  if (ok) {
    const int g8 = (lane >> 4) * 8;
#pragma unroll
    for (int i = 0; i < 8; ++i) {
      const int k0 = ((i < 4) ? (2 * i) : (16 + 2 * (i - 4))) + g8 + kbase;
      f[2 * i]     = rowp[k0];
      f[2 * i + 1] = rowp[k0 + 1];
    }
  }
  return f;
}

// ---------------- LDS staging (gfx1250 async copy path) ----------------
// Copies a workgroup-shared operand tile into LDS. 256-thread blocks.
DEVFN void stage_lds(void* sdst, const void* gsrc, int nbytes) {
#if defined(__gfx1250__)
  const unsigned lds0 = (unsigned)(uintptr_t)sdst;
  const unsigned long long g0 = (unsigned long long)(uintptr_t)gsrc;
  for (int i = (int)threadIdx.x * 16; i < nbytes; i += 256 * 16) {
    const unsigned l = lds0 + (unsigned)i;
    const unsigned long long g = g0 + (unsigned)i;
    asm volatile("global_load_async_to_lds_b128 %0, %1, off"
                 :: "v"(l), "v"(g) : "memory");
  }
  asm volatile("s_wait_asynccnt 0x0" ::: "memory");
#else
  for (int i = (int)threadIdx.x * 16; i < nbytes; i += 256 * 16)
    *(uint4*)((char*)sdst + i) = *(const uint4*)((const char*)gsrc + i);
#endif
}

// ---------------- packing kernels ----------------
__global__ __launch_bounds__(256) void pack_feats_k(
    const float* __restrict__ f0, const float* __restrict__ f1,
    _Float16* __restrict__ a0, _Float16* __restrict__ a1, float scl) {
  const int t = blockIdx.x * 256 + threadIdx.x;   // 2*4096*256 elements
  a0[t] = (_Float16)(f0[t] * scl);
  a1[t] = (_Float16)(f1[t] * scl);
}

__global__ __launch_bounds__(256) void pack_weights_k(
    const float* __restrict__ conv_w, const float* __restrict__ proj_w,
    const float* __restrict__ merge_w,
    _Float16* __restrict__ WcT, _Float16* __restrict__ PrT,
    _Float16* __restrict__ MtT, _Float16* __restrict__ MbT) {
  const int t = blockIdx.x * 256 + threadIdx.x;   // 32768 threads
  const int n = t >> 8, k = t & 255;              // [128,256] transposes
  WcT[t] = (_Float16)conv_w[(size_t)k * CF + n];
  PrT[t] = (_Float16)proj_w[(size_t)k * CF + n];
  if (t < CF * CF) {
    const int n2 = t >> 7, k2 = t & 127;
    MtT[t] = (_Float16)merge_w[(size_t)k2 * CF + n2];          // rows 0..127
    MbT[t] = (_Float16)merge_w[(size_t)(CF + k2) * CF + n2];   // rows 128..255
  }
}

// ---------------- sim GEMM: S[b,i,j] = scaled(q.rT) ----------------
__global__ __launch_bounds__(256) void sim_gemm_k(
    const _Float16* __restrict__ A0, const _Float16* __restrict__ A1,
    float* __restrict__ S) {
  const int b    = blockIdx.z;
  const int wave = threadIdx.x >> 5;
  const int lane = threadIdx.x & 31;
  const int wm = wave >> 1, wn = wave & 1;        // 4x2 waves -> 128x64 block
  const int row0 = blockIdx.x * 128 + wm * 32;
  const int col0 = blockIdx.y * 64 + wn * 32;
  const _Float16* A = A0 + (size_t)b * LSEQ * DIM;
  const _Float16* B = A1 + (size_t)b * LSEQ * DIM;
  v8f acc[2][2] = {};
#pragma unroll
  for (int kt = 0; kt < 8; ++kt) {
    const int k = kt * 32;
    if (kt < 7) {
      __builtin_prefetch(A + (size_t)row0 * DIM + k + 32, 0, 3);
      __builtin_prefetch(B + (size_t)col0 * DIM + k + 32, 0, 3);
    }
    v16h a0 = load_frag(A + (size_t)row0 * DIM + k, DIM, lane);
    v16h a1 = load_frag(A + (size_t)(row0 + 16) * DIM + k, DIM, lane);
    v16h b0 = load_frag(B + (size_t)col0 * DIM + k, DIM, lane);
    v16h b1 = load_frag(B + (size_t)(col0 + 16) * DIM + k, DIM, lane);
    acc[0][0] = wmma_f16(a0, b0, acc[0][0]);
    acc[0][1] = wmma_f16(a0, b1, acc[0][1]);
    acc[1][0] = wmma_f16(a1, b0, acc[1][0]);
    acc[1][1] = wmma_f16(a1, b1, acc[1][1]);
  }
  float* Sb = S + (size_t)b * LSEQ * LSEQ;
  const int n = lane & 15, hi = lane >> 4;
#pragma unroll
  for (int dm = 0; dm < 2; ++dm)
#pragma unroll
    for (int dn = 0; dn < 2; ++dn)
#pragma unroll
      for (int i = 0; i < 8; ++i)
        Sb[(size_t)(row0 + dm * 16 + i + 8 * hi) * LSEQ + col0 + dn * 16 + n] =
            acc[dm][dn][i];
}

// ---------------- softmax statistics (online) ----------------
__global__ __launch_bounds__(256) void row_stats_k(
    const float* __restrict__ S, float* __restrict__ Mr, float* __restrict__ Sr) {
  const int row = blockIdx.x;                       // b*L + i
  const float* p = S + (size_t)row * LSEQ;
  float m = -1e30f, s = 0.f;
  for (int j = threadIdx.x; j < LSEQ; j += 256) {
    const float x = p[j];
    const float mn = fmaxf(m, x);
    s = s * __expf(m - mn) + __expf(x - mn);
    m = mn;
  }
  __shared__ float sm[256], ss[256];
  sm[threadIdx.x] = m; ss[threadIdx.x] = s;
  __syncthreads();
  for (int off = 128; off; off >>= 1) {
    if ((int)threadIdx.x < off) {
      const float m2 = sm[threadIdx.x + off], s2 = ss[threadIdx.x + off];
      const float mn = fmaxf(sm[threadIdx.x], m2);
      ss[threadIdx.x] = ss[threadIdx.x] * __expf(sm[threadIdx.x] - mn) +
                        s2 * __expf(m2 - mn);
      sm[threadIdx.x] = mn;
    }
    __syncthreads();
  }
  if (threadIdx.x == 0) { Mr[row] = sm[0]; Sr[row] = ss[0]; }
}

// chunked column stats: grid.y = CCHUNK chunks of the i-range
__global__ __launch_bounds__(256) void col_stats_part_k(
    const float* __restrict__ S, float* __restrict__ Mp, float* __restrict__ Sp) {
  const int idx = blockIdx.x * 256 + threadIdx.x;   // b*L + j
  const int c = blockIdx.y;
  const int b = idx >> 12, j = idx & 4095;
  const float* p = S + (size_t)b * LSEQ * LSEQ + j +
                   (size_t)(c * (LSEQ / CCHUNK)) * LSEQ;
  float m = -1e30f, s = 0.f;
  for (int i = 0; i < LSEQ / CCHUNK; ++i) {
    const float x = p[(size_t)i * LSEQ];
    const float mn = fmaxf(m, x);
    s = s * __expf(m - mn) + __expf(x - mn);
    m = mn;
  }
  Mp[c * (BATCH * LSEQ) + idx] = m;
  Sp[c * (BATCH * LSEQ) + idx] = s;
}

__global__ __launch_bounds__(256) void col_stats_merge_k(
    const float* __restrict__ Mp, const float* __restrict__ Sp,
    float* __restrict__ Mc, float* __restrict__ Sc) {
  const int idx = blockIdx.x * 256 + threadIdx.x;
  float m = -1e30f, s = 0.f;
#pragma unroll
  for (int c = 0; c < CCHUNK; ++c) {
    const float m2 = Mp[c * (BATCH * LSEQ) + idx];
    const float s2 = Sp[c * (BATCH * LSEQ) + idx];
    const float mn = fmaxf(m, m2);
    s = s * __expf(m - mn) + s2 * __expf(m2 - mn);
    m = mn;
  }
  Mc[idx] = m; Sc[idx] = s;
}

// cm = exp(2S - Mr - Mc)/(Sr*Sc), in place; fused per-row max+argmax.
__global__ __launch_bounds__(256) void cm_transform_k(
    float* __restrict__ S, const float* __restrict__ Mr, const float* __restrict__ Sr,
    const float* __restrict__ Mc, const float* __restrict__ Sc,
    float* __restrict__ rowM, int* __restrict__ rowA) {
  const int row = blockIdx.x;
  const int b = row >> 12;
  float* p = S + (size_t)row * LSEQ;
  const float mr = Mr[row], inv_sr = 1.f / Sr[row];
  const float* mc = Mc + (size_t)b * LSEQ;
  const float* sc = Sc + (size_t)b * LSEQ;
  float best = -1.f; int bestj = 0;
  for (int j = threadIdx.x; j < LSEQ; j += 256) {
    const float x = p[j];
    const float v = __expf(x - mr) * inv_sr * __expf(x - mc[j]) / sc[j];
    p[j] = v;
    if (v > best) { best = v; bestj = j; }
  }
  __shared__ float sm[256]; __shared__ int sa[256];
  sm[threadIdx.x] = best; sa[threadIdx.x] = bestj;
  __syncthreads();
  for (int off = 128; off; off >>= 1) {
    if ((int)threadIdx.x < off && sm[threadIdx.x + off] > sm[threadIdx.x]) {
      sm[threadIdx.x] = sm[threadIdx.x + off];
      sa[threadIdx.x] = sa[threadIdx.x + off];
    }
    __syncthreads();
  }
  if (threadIdx.x == 0) { rowM[row] = sm[0]; rowA[row] = sa[0]; }
}

__global__ __launch_bounds__(256) void colmax_part_k(
    const float* __restrict__ CM, float* __restrict__ Xp) {
  const int idx = blockIdx.x * 256 + threadIdx.x;
  const int c = blockIdx.y;
  const int b = idx >> 12, j = idx & 4095;
  const float* p = CM + (size_t)b * LSEQ * LSEQ + j +
                   (size_t)(c * (LSEQ / CCHUNK)) * LSEQ;
  float m = 0.f;
  for (int i = 0; i < LSEQ / CCHUNK; ++i) m = fmaxf(m, p[(size_t)i * LSEQ]);
  Xp[c * (BATCH * LSEQ) + idx] = m;
}

__global__ __launch_bounds__(256) void colmax_merge_k(
    const float* __restrict__ Xp, float* __restrict__ colM) {
  const int idx = blockIdx.x * 256 + threadIdx.x;
  float m = 0.f;
#pragma unroll
  for (int c = 0; c < CCHUNK; ++c) m = fmaxf(m, Xp[c * (BATCH * LSEQ) + idx]);
  colM[idx] = m;
}

__global__ __launch_bounds__(256) void mutual_k(
    const float* __restrict__ rowM, const int* __restrict__ rowA,
    const float* __restrict__ colM, float* __restrict__ outJ,
    float* __restrict__ outV, int* __restrict__ jint) {
  const int row = blockIdx.x * 256 + threadIdx.x;   // 8192
  const int b = row >> 12;
  const int j = rowA[row];
  const float v = rowM[row];
  const bool mut = (v >= colM[b * LSEQ + j]);
  const int jid = mut ? j : 0;
  outJ[row] = (float)jid;
  outV[row] = mut ? 1.f : 0.f;
  jint[row] = jid;
}

// ---------------- conv1x1 + BN + Mish -> f16 g maps ----------------
__global__ __launch_bounds__(256) void conv_gemm_k(
    const float* __restrict__ x0, const float* __restrict__ x1,
    const _Float16* __restrict__ WcT, const float* __restrict__ cb,
    const float* __restrict__ gma, const float* __restrict__ bta,
    const float* __restrict__ mean, const float* __restrict__ var,
    _Float16* __restrict__ g) {
  __shared__ _Float16 sW[CF * DIM];          // 64 KB weight tile, reused 8x per wave
#if defined(USE_TDM)
  if (threadIdx.x < 32) {                    // wave 0 drives the Tensor Data Mover
    const unsigned lds = (unsigned)(uintptr_t)(void*)sW;
    const unsigned long long ga = (unsigned long long)(uintptr_t)WcT;
    v4u g0; v8i g1; v4i g2, g3;
    g0[0] = 1u;                                       // count=1, user descriptor
    g0[1] = lds;                                      // lds_addr
    g0[2] = (unsigned)ga;                             // global_addr[31:0]
    g0[3] = (unsigned)(ga >> 32) | 0x80000000u;       // global_addr[56:32] | type=2
    g1[0] = 0x00010000;          // workgroup_mask=0, data_size=1 (2 bytes)
    g1[1] = (int)0x80000000;     // tensor_dim0 = 32768 (low 16 bits -> [63:48])
    g1[2] = 0x00010000;          // tensor_dim0 hi=0, tensor_dim1 = 1
    g1[3] = (int)0x80000000;     // tile_dim0 = 32768 -> [127:112]
    g1[4] = 0; g1[5] = 0; g1[6] = 0; g1[7] = 0;
    g2[0] = g2[1] = g2[2] = g2[3] = 0;
    g3[0] = g3[1] = g3[2] = g3[3] = 0;
    __builtin_amdgcn_tensor_load_to_lds(g0, g1, g2, g3, 0);
    __builtin_amdgcn_s_wait_tensorcnt(0);
  }
#else
  stage_lds(sW, WcT, CF * DIM * 2);
#endif
  __syncthreads();

  const int z = blockIdx.z;                  // 0..3: map=z>>1, b=z&1
  const int mapi = z >> 1, b = z & 1;
  const float* x = (mapi ? x1 : x0) + (size_t)b * DIM * HW;
  const int wave = threadIdx.x >> 5, lane = threadIdx.x & 31;
  const int wm = wave >> 2, wn = wave & 3;   // 2x4 waves -> 64 pix x 128 ch
  const int pix0 = blockIdx.x * 64 + wm * 32;
  const int n0 = wn * 32;
  v8f acc[2][2] = {};
#pragma unroll
  for (int kt = 0; kt < 8; ++kt) {
    const int k = kt * 32;
    v16h a0 = load_frag_nchw(x, pix0, k, lane);
    v16h a1 = load_frag_nchw(x, pix0 + 16, k, lane);
    v16h b0 = load_frag(sW + (size_t)n0 * DIM + k, DIM, lane);
    v16h b1 = load_frag(sW + (size_t)(n0 + 16) * DIM + k, DIM, lane);
    acc[0][0] = wmma_f16(a0, b0, acc[0][0]);
    acc[0][1] = wmma_f16(a0, b1, acc[0][1]);
    acc[1][0] = wmma_f16(a1, b0, acc[1][0]);
    acc[1][1] = wmma_f16(a1, b1, acc[1][1]);
  }
  _Float16* gout = g + ((size_t)mapi * 2 + b) * HW * CF;
  const int hi = lane >> 4;
#pragma unroll
  for (int dn = 0; dn < 2; ++dn) {
    const int n = n0 + dn * 16 + (lane & 15);
    const float bias = cb[n];
    const float scal = gma[n] * rsqrtf(var[n] + 1e-5f);
    const float mu = mean[n], bt = bta[n];
#pragma unroll
    for (int dm = 0; dm < 2; ++dm)
#pragma unroll
      for (int i = 0; i < 8; ++i) {
        const int m = pix0 + dm * 16 + i + 8 * hi;
        float y = acc[dm][dn][i] + bias;
        y = (y - mu) * scal + bt;
        const float sp = (y > 20.f) ? y : log1pf(__expf(y));
        gout[(size_t)m * CF + n] = (_Float16)(y * tanhf(sp));
      }
  }
}

// ---------------- center path ----------------
__global__ __launch_bounds__(256) void center_pack_k(
    const float* __restrict__ f0, const float* __restrict__ f1,
    const int* __restrict__ jint, _Float16* __restrict__ Cmat) {
  const int t = blockIdx.x * 256 + threadIdx.x;  // 16384*256
  const int r = t >> 8, c = t & 255;
  float v;
  if (r < 8192) {
    v = f0[(size_t)r * DIM + c];
  } else {
    const int r2 = r - 8192;
    const int b = r2 >> 12, l = r2 & 4095;
    const int j = jint[b * LSEQ + l];
    v = f1[((size_t)b * LSEQ + j) * DIM + c];
  }
  Cmat[t] = (_Float16)v;
}

__global__ __launch_bounds__(256) void proj_gemm_k(
    const _Float16* __restrict__ Cmat, const _Float16* __restrict__ PrT,
    const float* __restrict__ pb, _Float16* __restrict__ P) {
  __shared__ _Float16 sB[CF * DIM];
  stage_lds(sB, PrT, CF * DIM * 2);
  __syncthreads();
  const int wave = threadIdx.x >> 5, lane = threadIdx.x & 31;
  const int wm = wave >> 2, wn = wave & 3;
  const int row0 = blockIdx.x * 64 + wm * 32;
  const int n0 = wn * 32;
  v8f acc[2][2] = {};
#pragma unroll
  for (int kt = 0; kt < 8; ++kt) {
    const int k = kt * 32;
    v16h a0 = load_frag(Cmat + (size_t)row0 * DIM + k, DIM, lane);
    v16h a1 = load_frag(Cmat + (size_t)(row0 + 16) * DIM + k, DIM, lane);
    v16h b0 = load_frag(sB + (size_t)n0 * DIM + k, DIM, lane);
    v16h b1 = load_frag(sB + (size_t)(n0 + 16) * DIM + k, DIM, lane);
    acc[0][0] = wmma_f16(a0, b0, acc[0][0]);
    acc[0][1] = wmma_f16(a0, b1, acc[0][1]);
    acc[1][0] = wmma_f16(a1, b0, acc[1][0]);
    acc[1][1] = wmma_f16(a1, b1, acc[1][1]);
  }
  const int hi = lane >> 4;
#pragma unroll
  for (int dn = 0; dn < 2; ++dn) {
    const int n = n0 + dn * 16 + (lane & 15);
    const float bias = pb[n];
#pragma unroll
    for (int dm = 0; dm < 2; ++dm)
#pragma unroll
      for (int i = 0; i < 8; ++i) {
        const int m = row0 + dm * 16 + i + 8 * hi;
        P[(size_t)m * CF + n] = (_Float16)(acc[dm][dn][i] + bias);
      }
  }
}

__global__ __launch_bounds__(256) void cc_gemm_k(
    const _Float16* __restrict__ P, const _Float16* __restrict__ MbT,
    const float* __restrict__ mb, float* __restrict__ CC) {
  __shared__ _Float16 sB[CF * CF];
  stage_lds(sB, MbT, CF * CF * 2);
  __syncthreads();
  const int wave = threadIdx.x >> 5, lane = threadIdx.x & 31;
  const int wm = wave >> 2, wn = wave & 3;
  const int row0 = blockIdx.x * 64 + wm * 32;
  const int n0 = wn * 32;
  v8f acc[2][2] = {};
#pragma unroll
  for (int kt = 0; kt < 4; ++kt) {
    const int k = kt * 32;
    v16h a0 = load_frag(P + (size_t)row0 * CF + k, CF, lane);
    v16h a1 = load_frag(P + (size_t)(row0 + 16) * CF + k, CF, lane);
    v16h b0 = load_frag(sB + (size_t)n0 * CF + k, CF, lane);
    v16h b1 = load_frag(sB + (size_t)(n0 + 16) * CF + k, CF, lane);
    acc[0][0] = wmma_f16(a0, b0, acc[0][0]);
    acc[0][1] = wmma_f16(a0, b1, acc[0][1]);
    acc[1][0] = wmma_f16(a1, b0, acc[1][0]);
    acc[1][1] = wmma_f16(a1, b1, acc[1][1]);
  }
  const int hi = lane >> 4;
#pragma unroll
  for (int dn = 0; dn < 2; ++dn) {
    const int n = n0 + dn * 16 + (lane & 15);
    const float bias = mb[n];
#pragma unroll
    for (int dm = 0; dm < 2; ++dm)
#pragma unroll
      for (int i = 0; i < 8; ++i) {
        const int m = row0 + dm * 16 + i + 8 * hi;
        CC[(size_t)m * CF + n] = acc[dm][dn][i] + bias;
      }
  }
}

// ---------------- merge GEMM with unfold gather ----------------
DEVFN const _Float16* merge_row_ptr(int rr, const int* __restrict__ jint,
                                    const _Float16* __restrict__ g, bool& ok) {
  const int r = rr / WW, ww = rr - r * WW;
  int b, patch;
  const _Float16* base;
  if (r < 8192) {
    b = r >> 12; patch = r & 4095; base = g;
  } else {
    const int r2 = r - 8192;
    b = r2 >> 12;
    const int l = r2 & 4095;
    patch = jint[b * LSEQ + l];
    base = g + (size_t)2 * HW * CF;
  }
  const int py = patch >> 6, px = patch & 63;
  const int wy = ww / 5, wx = ww - wy * 5;
  const int y = py * 4 - 2 + wy;
  const int x = px * 4 - 2 + wx;
  ok = (y >= 0) & (y < HF) & (x >= 0) & (x < HF);
  return base + ((size_t)b * HW + y * HF + x) * CF;
}

__global__ __launch_bounds__(256) void merge_gemm_k(
    const _Float16* __restrict__ g, const int* __restrict__ jint,
    const _Float16* __restrict__ MtT, const float* __restrict__ CC,
    float* __restrict__ out) {
  __shared__ _Float16 sB[CF * CF];
  stage_lds(sB, MtT, CF * CF * 2);
  __syncthreads();
  const int wave = threadIdx.x >> 5, lane = threadIdx.x & 31;
  const int wm = wave >> 2, wn = wave & 3;      // 64 rows x 128 cols per block
  const int row0 = blockIdx.x * 64 + wm * 32;
  const int n0 = wn * 32;

  bool ok0, ok1;
  const _Float16* rp0 = merge_row_ptr(row0 + (lane & 15), jint, g, ok0);
  const _Float16* rp1 = merge_row_ptr(row0 + 16 + (lane & 15), jint, g, ok1);

  v8f acc[2][2] = {};
#pragma unroll
  for (int kt = 0; kt < 4; ++kt) {
    const int k = kt * 32;
    v16h a0 = load_frag_row(rp0, ok0, k, lane);
    v16h a1 = load_frag_row(rp1, ok1, k, lane);
    v16h b0 = load_frag(sB + (size_t)n0 * CF + k, CF, lane);
    v16h b1 = load_frag(sB + (size_t)(n0 + 16) * CF + k, CF, lane);
    acc[0][0] = wmma_f16(a0, b0, acc[0][0]);
    acc[0][1] = wmma_f16(a0, b1, acc[0][1]);
    acc[1][0] = wmma_f16(a1, b0, acc[1][0]);
    acc[1][1] = wmma_f16(a1, b1, acc[1][1]);
  }
  const int hi = lane >> 4;
  const float LN1E4_DIV = 9.210340371976184f / 128.f;
#pragma unroll
  for (int dn = 0; dn < 2; ++dn) {
    const int n = n0 + dn * 16 + (lane & 15);
    const float e = __expf(-(float)(n & ~1) * LN1E4_DIV);
#pragma unroll
    for (int dm = 0; dm < 2; ++dm)
#pragma unroll
      for (int i = 0; i < 8; ++i) {
        const int rr = row0 + dm * 16 + i + 8 * hi;
        const int r = rr / WW, ww = rr - r * WW;
        const float ang = (float)ww * e;
        const float pe = (n & 1) ? __cosf(ang) : __sinf(ang);
        out[(size_t)rr * CF + n] = acc[dm][dn][i] + CC[(size_t)r * CF + n] + pe;
      }
  }
}

// ---------------- launch ----------------
extern "C" void kernel_launch(void* const* d_in, const int* in_sizes, int n_in,
                              void* d_out, int out_size, void* d_ws, size_t ws_size,
                              hipStream_t stream) {
  const float* feat0   = (const float*)d_in[0];
  const float* feat1   = (const float*)d_in[1];
  const float* fm0     = (const float*)d_in[2];
  const float* fm1     = (const float*)d_in[3];
  const float* conv_w  = (const float*)d_in[4];
  const float* conv_b  = (const float*)d_in[5];
  const float* bn_g    = (const float*)d_in[6];
  const float* bn_b    = (const float*)d_in[7];
  const float* bn_m    = (const float*)d_in[8];
  const float* bn_v    = (const float*)d_in[9];
  const float* proj_w  = (const float*)d_in[10];
  const float* proj_b  = (const float*)d_in[11];
  const float* merge_w = (const float*)d_in[12];
  const float* merge_b = (const float*)d_in[13];

  float* out = (float*)d_out;
  float* out_ld = out;
  float* out_cm = out + OUT_CM_OFF;
  float* out_j  = out + OUT_J_OFF;
  float* out_v  = out + OUT_V_OFF;

  char* w = (char*)d_ws;
  size_t off = 0;
  auto alloc = [&](size_t bytes) {
    void* p = w + off;
    off += (bytes + 255) & ~(size_t)255;
    return p;
  };
  _Float16* A0h  = (_Float16*)alloc((size_t)BATCH * LSEQ * DIM * 2);
  _Float16* A1h  = (_Float16*)alloc((size_t)BATCH * LSEQ * DIM * 2);
  _Float16* WcT  = (_Float16*)alloc((size_t)CF * DIM * 2);
  _Float16* PrT  = (_Float16*)alloc((size_t)CF * DIM * 2);
  _Float16* MtT  = (_Float16*)alloc((size_t)CF * CF * 2);
  _Float16* MbT  = (_Float16*)alloc((size_t)CF * CF * 2);
  float* Mr      = (float*)alloc((size_t)BATCH * LSEQ * 4);
  float* Sr      = (float*)alloc((size_t)BATCH * LSEQ * 4);
  float* Mc      = (float*)alloc((size_t)BATCH * LSEQ * 4);
  float* Sc      = (float*)alloc((size_t)BATCH * LSEQ * 4);
  float* rowM    = (float*)alloc((size_t)BATCH * LSEQ * 4);
  int*   rowA    = (int*)alloc((size_t)BATCH * LSEQ * 4);
  float* colM    = (float*)alloc((size_t)BATCH * LSEQ * 4);
  int*   jint    = (int*)alloc((size_t)BATCH * LSEQ * 4);
  float* Mp      = (float*)alloc((size_t)CCHUNK * BATCH * LSEQ * 4);
  float* Sp      = (float*)alloc((size_t)CCHUNK * BATCH * LSEQ * 4);
  float* Xp      = (float*)alloc((size_t)CCHUNK * BATCH * LSEQ * 4);
  _Float16* gbuf = (_Float16*)alloc((size_t)4 * HW * CF * 2);
  _Float16* Cmat = (_Float16*)alloc((size_t)NMATCH * DIM * 2);
  _Float16* Pmat = (_Float16*)alloc((size_t)NMATCH * CF * 2);
  float* CC      = (float*)alloc((size_t)NMATCH * CF * 4);
  (void)ws_size; (void)in_sizes; (void)n_in; (void)out_size;

  const float scl = 1.0f / sqrtf((float)DIM * (float)DIM * 0.1f);

  // --- coarse branch ---
  pack_feats_k<<<dim3((BATCH * LSEQ * DIM) / 256), 256, 0, stream>>>(
      feat0, feat1, A0h, A1h, scl);
  pack_weights_k<<<dim3((CF * DIM) / 256), 256, 0, stream>>>(
      conv_w, proj_w, merge_w, WcT, PrT, MtT, MbT);
  sim_gemm_k<<<dim3(LSEQ / 128, LSEQ / 64, BATCH), 256, 0, stream>>>(
      A0h, A1h, out_cm);
  row_stats_k<<<dim3(BATCH * LSEQ), 256, 0, stream>>>(out_cm, Mr, Sr);
  col_stats_part_k<<<dim3((BATCH * LSEQ) / 256, CCHUNK), 256, 0, stream>>>(
      out_cm, Mp, Sp);
  col_stats_merge_k<<<dim3((BATCH * LSEQ) / 256), 256, 0, stream>>>(Mp, Sp, Mc, Sc);
  cm_transform_k<<<dim3(BATCH * LSEQ), 256, 0, stream>>>(
      out_cm, Mr, Sr, Mc, Sc, rowM, rowA);
  colmax_part_k<<<dim3((BATCH * LSEQ) / 256, CCHUNK), 256, 0, stream>>>(out_cm, Xp);
  colmax_merge_k<<<dim3((BATCH * LSEQ) / 256), 256, 0, stream>>>(Xp, colM);
  mutual_k<<<dim3((BATCH * LSEQ) / 256), 256, 0, stream>>>(
      rowM, rowA, colM, out_j, out_v, jint);

  // --- fine branch ---
  conv_gemm_k<<<dim3(HW / 64, 1, 4), 256, 0, stream>>>(
      fm0, fm1, WcT, conv_b, bn_g, bn_b, bn_m, bn_v, gbuf);
  center_pack_k<<<dim3((NMATCH * DIM) / 256), 256, 0, stream>>>(
      feat0, feat1, jint, Cmat);
  proj_gemm_k<<<dim3(NMATCH / 64), 256, 0, stream>>>(Cmat, PrT, proj_b, Pmat);
  cc_gemm_k<<<dim3(NMATCH / 64), 256, 0, stream>>>(Pmat, MbT, merge_b, CC);
  merge_gemm_k<<<dim3(MROWS / 64), 256, 0, stream>>>(gbuf, jint, MtT, CC, out_ld);
}
